// MultiHeadAttention_80745385164993
// MI455X (gfx1250) — matrix operand order repaired and checked by reading, not compiled, not run on hardware
//
#include <hip/hip_runtime.h>

typedef __attribute__((ext_vector_type(16))) _Float16 v16h;
typedef __attribute__((ext_vector_type(8)))  float    v8f;
typedef __attribute__((ext_vector_type(4)))  unsigned int u32x4;
typedef __attribute__((ext_vector_type(8)))  int      i32x8;
typedef __attribute__((ext_vector_type(4)))  int      i32x4;

#define NB    2
#define LSEQ  2048
#define NH    16
#define DH    64
#define DM    1024

// Tensor Data Mover availability / arity detection
#if defined(__has_builtin)
#if __has_builtin(__builtin_amdgcn_tensor_load_to_lds)
#define USE_TDM 1
#endif
#endif
#if defined(__has_include)
#if __has_include(<hip/amd_detail/amd_gfx1250_TDM.h>)
#define TDM_6ARG 1   // therock-10.0 headers -> 6-arg builtin form
#endif
#endif

// ---------------------------------------------------------------------------
// WMMA wrapper: D = A(16x32 f16) * B(32x16 f16) + C(16x16 f32)
// ---------------------------------------------------------------------------
__device__ __forceinline__ v8f wmma16(v16h a, v16h b, v8f c) {
  return __builtin_amdgcn_wmma_f32_16x16x32_f16(
      /*neg_a=*/false, a, /*neg_b=*/false, b,
      /*c_mod=*/(short)0, c, /*reuse_a=*/false, /*reuse_b=*/false);
}

// ---------------------------------------------------------------------------
// Fragment loaders (wave32; A/B use contiguous k order, same in both, not the ISA table order)
// A 16x32 (MxK): lane l -> M = l&15; halves i -> K = k0 + i + 16*(l>=16)
// B 32x16 (KxN): lane l -> N = l&15; halves i -> K = k0 + i + 16*(l>=16)
// C/D 16x16:     lane l -> N = l&15; vgpr j -> M = j + 8*(l>=16)
// ---------------------------------------------------------------------------
__device__ __forceinline__ v16h load_a_h(const _Float16* base, int ld, int lane, int k0) {
  int m  = lane & 15;
  int hi = (lane >> 4) & 1;
  const _Float16* row = base + (size_t)m * ld;
  v16h a;
#pragma unroll
  for (int i = 0; i < 16; ++i) {
    int k = k0 + i + hi * 16;
    a[i] = row[k];
  }
  return a;
}

__device__ __forceinline__ v16h load_a_f32(const float* base, int ld, int lane, int k0) {
  int m  = lane & 15;
  int hi = (lane >> 4) & 1;
  const float* row = base + (size_t)m * ld;
  v16h a;
#pragma unroll
  for (int i = 0; i < 16; ++i) {
    int k = k0 + i + hi * 16;
    a[i] = (_Float16)row[k];
  }
  return a;
}

// B[k][n] = src[n][k]  (reads rows of a row-major src -> implicit transpose)
__device__ __forceinline__ v16h load_bT_h(const _Float16* base, int ld, int lane,
                                          int k0, int n0) {
  int n    = n0 + (lane & 15);
  int koff = k0 + ((lane >> 4) & 1) * 16;
  const _Float16* p = base + (size_t)n * ld + koff;
  v16h b;
#pragma unroll
  for (int i = 0; i < 16; ++i) b[i] = p[i];
  return b;
}

__device__ __forceinline__ v16h load_bT_f32(const float* base, int ld, int lane,
                                            int k0, int n0) {
  int n    = n0 + (lane & 15);
  int koff = k0 + ((lane >> 4) & 1) * 16;
  const float* p = base + (size_t)n * ld + koff;
  v16h b;
#pragma unroll
  for (int i = 0; i < 16; ++i) b[i] = (_Float16)p[i];
  return b;
}

// B[k][n] = src[k][n]  (direct layout, strided reads)
__device__ __forceinline__ v16h load_bD_h(const _Float16* base, int ld, int lane,
                                          int k0, int n0) {
  int n    = n0 + (lane & 15);
  int koff = k0 + ((lane >> 4) & 1) * 16;
  v16h b;
#pragma unroll
  for (int i = 0; i < 16; ++i) b[i] = base[(size_t)(koff + i) * ld + n];
  return b;
}

// ---------------------------------------------------------------------------
// TDM: async-load one 64-row x 128-byte tile (row stride 128B in memory)
// into LDS with +32B padding per row -> matches _Float16 [64][80] layout.
// D# group1 is fully compile-time constant:
//   data_size=8B (code 3), pad_enable=1, pad_interval=128B (code 4),
//   pad_amount=32B (code 7), tensor_dim0=16 units, tensor_dim1=64,
//   tile_dim0=16, tile_dim1=64, tensor_dim0_stride=16 units.
// ---------------------------------------------------------------------------
#if defined(USE_TDM)
__device__ __forceinline__ void tdm_load_tile64(const void* gptr, unsigned lds_off) {
  unsigned long long ga = (unsigned long long)(uintptr_t)gptr;
  u32x4 g0;
  g0[0] = 1u;                                              // count=1, user mode
  g0[1] = lds_off;                                         // LDS byte address
  g0[2] = (unsigned)ga;                                    // global_addr[31:0]
  g0[3] = ((unsigned)(ga >> 32) & 0x01FFFFFFu) | 0x80000000u; // addr[56:32]|type=2
  i32x8 g1;
  g1[0] = 0x0F130000;   // data_size=3(8B) | pad_enable | interval=4 | amount=7
  g1[1] = 16 << 16;     // tensor_dim0[15:0] in bits 63:48
  g1[2] = 64 << 16;     // tensor_dim1[15:0] in bits 95:80
  g1[3] = 16 << 16;     // tile_dim0 in bits 127:112
  g1[4] = 64;           // tile_dim1 in bits 143:128
  g1[5] = 16;           // tensor_dim0_stride[31:0] in bits 191:160
  g1[6] = 0;
  g1[7] = 0;
  i32x4 z4 = {0, 0, 0, 0};
#if defined(TDM_6ARG)
  i32x8 z8 = {0, 0, 0, 0, 0, 0, 0, 0};
  __builtin_amdgcn_tensor_load_to_lds(g0, g1, z4, z4, z8, 0);
#else
  __builtin_amdgcn_tensor_load_to_lds(g0, g1, z4, z4, 0);
#endif
}
#endif

// ---------------------------------------------------------------------------
// Kernel 1: per-head linear projection  out = x @ W^T + bias
// rows = (b,l,h) flattened = NB*LSEQ*NH; K = N = 64.
// Output stored head-major: [B, H, L, 64] (f16).
// ---------------------------------------------------------------------------
__global__ __launch_bounds__(256) void proj_qkv(const float* __restrict__ X,
                                                const float* __restrict__ W,
                                                const float* __restrict__ bias,
                                                _Float16* __restrict__ Out) {
  int lane = threadIdx.x & 31;
  int wave = threadIdx.x >> 5;
  int tile = blockIdx.x * 8 + wave;   // 4096 tiles of 16 rows
  int r0   = tile * 16;
  int hi   = (lane >> 4) & 1;

  v16h a0 = load_a_f32(X + (size_t)r0 * DH, DH, lane, 0);
  v16h a1 = load_a_f32(X + (size_t)r0 * DH, DH, lane, 32);

#pragma unroll
  for (int nt = 0; nt < 4; ++nt) {
    v16h b0 = load_bT_f32(W, DH, lane, 0,  nt * 16);
    v16h b1 = load_bT_f32(W, DH, lane, 32, nt * 16);
    v8f c = {0.f, 0.f, 0.f, 0.f, 0.f, 0.f, 0.f, 0.f};
    c = wmma16(a0, b0, c);
    c = wmma16(a1, b1, c);

    int n   = nt * 16 + (lane & 15);
    float bv = bias[n];
#pragma unroll
    for (int j = 0; j < 8; ++j) {
      int m   = j + hi * 8;
      int row = r0 + m;                  // row = (b*LSEQ + l)*NH + h
      int bb  = row >> 15;
      int rem = row & 32767;
      int ll  = rem >> 4;
      int hh  = rem & 15;
      size_t o = ((((size_t)bb * NH + hh) * LSEQ) + ll) * DH + n;
      Out[o] = (_Float16)(c[j] + bv);
    }
  }
}

// ---------------------------------------------------------------------------
// Kernel 2: fp32 -> f16 weight convert (for Wo)
// ---------------------------------------------------------------------------
__global__ void cvt_f32_f16(const float* __restrict__ src,
                            _Float16* __restrict__ dst, int n) {
  int i = blockIdx.x * 256 + threadIdx.x;
  if (i < n) dst[i] = (_Float16)src[i];
}

// ---------------------------------------------------------------------------
// Kernel 3: flash attention per (b,h). Block = 8 waves; each wave owns a
// 16-row Q tile. K/V tiles double-buffered in LDS, staged by the TDM
// (tensor_load_to_lds + s_wait_tensorcnt) issued from wave 0, overlapping
// DMA of tile i+1 with WMMA compute on tile i.
// ---------------------------------------------------------------------------
__global__ __launch_bounds__(256) void attn_kernel(const _Float16* __restrict__ Qh,
                                                   const _Float16* __restrict__ Kh,
                                                   const _Float16* __restrict__ Vh,
                                                   _Float16* __restrict__ Oh) {
  __shared__ alignas(16) _Float16 Ks[2][64][80];    // padded rows (TDM pad)
  __shared__ alignas(16) _Float16 Vs[2][64][80];
  __shared__ alignas(16) _Float16 Ps[8][16][72];    // per-wave P re-layout

  int lane = threadIdx.x & 31;
  int wave = threadIdx.x >> 5;
  int hi   = (lane >> 4) & 1;

  int bh = blockIdx.x;            // b*NH + h
  int b_ = bh >> 4;
  int h_ = bh & 15;
  const _Float16* Q = Qh + (size_t)bh * LSEQ * DH;
  const _Float16* K = Kh + (size_t)bh * LSEQ * DH;
  const _Float16* V = Vh + (size_t)bh * LSEQ * DH;

  int q0 = blockIdx.y * 128 + wave * 16;

  v16h qa0 = load_a_h(Q + (size_t)q0 * DH, DH, lane, 0);
  v16h qa1 = load_a_h(Q + (size_t)q0 * DH, DH, lane, 32);

  v8f acc[4];
  float mrow[8], lrow[8];
#pragma unroll
  for (int f = 0; f < 4; ++f) {
    v8f z = {0.f, 0.f, 0.f, 0.f, 0.f, 0.f, 0.f, 0.f};
    acc[f] = z;
  }
#pragma unroll
  for (int j = 0; j < 8; ++j) { mrow[j] = -1e30f; lrow[j] = 0.f; }

  const float scale = 0.03125f;   // 1/sqrt(1024)
  const int NIT = LSEQ / 64;      // 32 key tiles

#if defined(USE_TDM)
  if (wave == 0) {
    tdm_load_tile64(K, (unsigned)(uintptr_t)&Ks[0][0][0]);
    tdm_load_tile64(V, (unsigned)(uintptr_t)&Vs[0][0][0]);
  }
#endif

  for (int it = 0; it < NIT; ++it) {
    int cur = it & 1;
    __syncthreads();   // all waves done with buffer (it-1)&1 == (it+1)&1
#if defined(USE_TDM)
    if (wave == 0) {
      if (it + 1 < NIT) {
        int nb = (it + 1) & 1;
        const _Float16* knext = K + (size_t)(it + 1) * 64 * DH;
        const _Float16* vnext = V + (size_t)(it + 1) * 64 * DH;
        tdm_load_tile64(knext, (unsigned)(uintptr_t)&Ks[nb][0][0]);
        tdm_load_tile64(vnext, (unsigned)(uintptr_t)&Vs[nb][0][0]);
        // TDM ops complete in order: <=2 outstanding => tile `it` is in LDS
        __builtin_amdgcn_s_wait_tensorcnt(2);
      } else {
        __builtin_amdgcn_s_wait_tensorcnt(0);
      }
    }
#else
    // fallback: cooperative staging (64 x 64 f16 per tensor, 16B vectors)
#pragma unroll
    for (int p = 0; p < 2; ++p) {
      int idx = p * 256 + threadIdx.x;
      int row = idx >> 3;
      int seg = idx & 7;
      *(uint4*)&Ks[cur][row][seg * 8] =
          *(const uint4*)(K + (size_t)(it * 64 + row) * DH + seg * 8);
      *(uint4*)&Vs[cur][row][seg * 8] =
          *(const uint4*)(V + (size_t)(it * 64 + row) * DH + seg * 8);
    }
#endif
    __syncthreads();

    // S = Q * K^T  (16 x 64), B[k][n] = Ks[n][k]
    v8f s[4];
#pragma unroll
    for (int nt = 0; nt < 4; ++nt) {
      v16h kb0 = load_bT_h(&Ks[cur][0][0], 80, lane, 0,  nt * 16);
      v16h kb1 = load_bT_h(&Ks[cur][0][0], 80, lane, 32, nt * 16);
      v8f c = {0.f, 0.f, 0.f, 0.f, 0.f, 0.f, 0.f, 0.f};
      c = wmma16(qa0, kb0, c);
      c = wmma16(qa1, kb1, c);
      s[nt] = c;
    }

    // scale + online softmax (rows 0-7 in lanes 0-15, rows 8-15 in 16-31)
#pragma unroll
    for (int nt = 0; nt < 4; ++nt)
#pragma unroll
      for (int j = 0; j < 8; ++j) s[nt][j] *= scale;

#pragma unroll
    for (int j = 0; j < 8; ++j) {
      float t = fmaxf(fmaxf(s[0][j], s[1][j]), fmaxf(s[2][j], s[3][j]));
      t = fmaxf(t, __shfl_xor(t, 1));
      t = fmaxf(t, __shfl_xor(t, 2));
      t = fmaxf(t, __shfl_xor(t, 4));
      t = fmaxf(t, __shfl_xor(t, 8));
      float mnew = fmaxf(mrow[j], t);
      float cr   = __expf(mrow[j] - mnew);
      mrow[j] = mnew;
      lrow[j] *= cr;
#pragma unroll
      for (int f = 0; f < 4; ++f) acc[f][j] *= cr;
    }

    float psum[8];
#pragma unroll
    for (int j = 0; j < 8; ++j) psum[j] = 0.f;
#pragma unroll
    for (int nt = 0; nt < 4; ++nt) {
      int ccol = nt * 16 + (lane & 15);
#pragma unroll
      for (int j = 0; j < 8; ++j) {
        float p = __expf(s[nt][j] - mrow[j]);
        psum[j] += p;
        Ps[wave][j + hi * 8][ccol] = (_Float16)p;
      }
    }
#pragma unroll
    for (int j = 0; j < 8; ++j) {
      float t = psum[j];
      t += __shfl_xor(t, 1);
      t += __shfl_xor(t, 2);
      t += __shfl_xor(t, 4);
      t += __shfl_xor(t, 8);
      lrow[j] += t;
    }

    __syncthreads();   // make P (C-layout) visible for A-layout reload

    // O += P * V : A = P (16x64 over key dim), B[k][d] = Vs[k][d]
    v16h pa0 = load_a_h(&Ps[wave][0][0], 72, lane, 0);
    v16h pa1 = load_a_h(&Ps[wave][0][0], 72, lane, 32);
#pragma unroll
    for (int f = 0; f < 4; ++f) {
      v16h vb0 = load_bD_h(&Vs[cur][0][0], 80, lane, 0,  f * 16);
      v16h vb1 = load_bD_h(&Vs[cur][0][0], 80, lane, 32, f * 16);
      acc[f] = wmma16(pa0, vb0, acc[f]);
      acc[f] = wmma16(pa1, vb1, acc[f]);
    }
  }

  // epilogue: O /= l, store concat-head layout [B, L, H*64] (f16)
#pragma unroll
  for (int f = 0; f < 4; ++f) {
    int d = f * 16 + (lane & 15);
#pragma unroll
    for (int j = 0; j < 8; ++j) {
      int m = j + hi * 8;
      float o = acc[f][j] / lrow[j];
      Oh[((size_t)(b_ * LSEQ + q0 + m)) * DM + h_ * DH + d] = (_Float16)o;
    }
  }
}

// ---------------------------------------------------------------------------
// Kernel 4: output projection  out = O @ Wo^T + bo   (4096 x 1024 x 1024)
// One wave per 16x64 output tile; GL2 prefetch of next K-panel.
// ---------------------------------------------------------------------------
__global__ __launch_bounds__(256) void out_proj(const _Float16* __restrict__ Oh,
                                                const _Float16* __restrict__ Woh,
                                                const float* __restrict__ bo,
                                                float* __restrict__ out) {
  int lane = threadIdx.x & 31;
  int wave = threadIdx.x >> 5;
  int tile = blockIdx.x * 8 + wave;       // 4096 tiles
  int mt = tile >> 4;
  int nt = tile & 15;
  int r0 = mt * 16;
  int n0 = nt * 64;
  int hi = (lane >> 4) & 1;

  v8f acc[4];
#pragma unroll
  for (int f = 0; f < 4; ++f) {
    v8f z = {0.f, 0.f, 0.f, 0.f, 0.f, 0.f, 0.f, 0.f};
    acc[f] = z;
  }

  for (int k0 = 0; k0 < DM; k0 += 32) {
    if (k0 + 32 < DM) {
      // warm GL2 for the next K-panel (emits global_prefetch_b8)
      __builtin_prefetch((const void*)(Oh + (size_t)(r0 + (lane & 15)) * DM + k0 + 32), 0, 1);
      __builtin_prefetch((const void*)(Woh + (size_t)(n0 + (lane & 15)) * DM + k0 + 32), 0, 1);
    }
    v16h a = load_a_h(Oh + (size_t)r0 * DM, DM, lane, k0);
#pragma unroll
    for (int f = 0; f < 4; ++f) {
      v16h b = load_bT_h(Woh, DM, lane, k0, n0 + f * 16);  // B[k][n] = Wo[n][k]
      acc[f] = wmma16(a, b, acc[f]);
    }
  }

#pragma unroll
  for (int f = 0; f < 4; ++f) {
    int n = n0 + f * 16 + (lane & 15);
    float bb = bo[n];
#pragma unroll
    for (int j = 0; j < 8; ++j) {
      int m = j + hi * 8;
      out[(size_t)(r0 + m) * DM + n] = acc[f][j] + bb;
    }
  }
}

// ---------------------------------------------------------------------------
// Host launcher
// ---------------------------------------------------------------------------
extern "C" void kernel_launch(void* const* d_in, const int* in_sizes, int n_in,
                              void* d_out, int out_size, void* d_ws, size_t ws_size,
                              hipStream_t stream) {
  (void)in_sizes; (void)n_in; (void)out_size; (void)ws_size;

  const float* values = (const float*)d_in[0];
  const float* keys   = (const float*)d_in[1];
  const float* query  = (const float*)d_in[2];
  const float* Wq = (const float*)d_in[3];
  const float* bq = (const float*)d_in[4];
  const float* Wk = (const float*)d_in[5];
  const float* bk = (const float*)d_in[6];
  const float* Wv = (const float*)d_in[7];
  const float* bv = (const float*)d_in[8];
  const float* Wo = (const float*)d_in[9];
  const float* bo = (const float*)d_in[10];
  float* outp = (float*)d_out;

  char* ws = (char*)d_ws;
  _Float16* Qh  = (_Float16*)(ws);                          // 8 MiB
  _Float16* Kh  = (_Float16*)(ws + (size_t)(8u  << 20));    // 8 MiB
  _Float16* Vh  = (_Float16*)(ws + (size_t)(16u << 20));    // 8 MiB
  _Float16* Oh  = (_Float16*)(ws + (size_t)(24u << 20));    // 8 MiB
  _Float16* Woh = (_Float16*)(ws + (size_t)(32u << 20));    // 2 MiB

  proj_qkv<<<512, 256, 0, stream>>>(query,  Wq, bq, Qh);
  proj_qkv<<<512, 256, 0, stream>>>(keys,   Wk, bk, Kh);
  proj_qkv<<<512, 256, 0, stream>>>(values, Wv, bv, Vh);

  cvt_f32_f16<<<(DM * DM + 255) / 256, 256, 0, stream>>>(Wo, Woh, DM * DM);

  attn_kernel<<<dim3(NB * NH, LSEQ / 128), 256, 0, stream>>>(Qh, Kh, Vh, Oh);

  out_proj<<<512, 256, 0, stream>>>(Oh, Woh, bo, outp);
}
